// CustomMoELayer_67826123538669
// MI455X (gfx1250) — matrix operand order repaired and checked by reading, not compiled, not run on hardware
//
#include <hip/hip_runtime.h>
#include <cstdint>

#define T_TOK 8192
#define HID   1024
#define IDIM  4096
#define NEXP  8

#define MT      32      // rows per block
#define ICHUNK  256     // I-chunk per iteration (16 waves x 16)
#define NWAVE   16      // 512 threads

typedef __attribute__((ext_vector_type(16))) __bf16 bf16x16;
typedef __attribute__((ext_vector_type(8)))  float  f32x8;

union ABFrag {
  unsigned int u[8];
  uint4        q[2];
  bf16x16      v;
};

__device__ __forceinline__ unsigned short f2bf(float f) {
  unsigned int u = __float_as_uint(f);
  u += 0x7FFFu + ((u >> 16) & 1u);            // round-to-nearest-even
  return (unsigned short)(u >> 16);
}
__device__ __forceinline__ unsigned int pack_bf2(float lo, float hi) {
  return (unsigned int)f2bf(lo) | ((unsigned int)f2bf(hi) << 16);
}

// ---------------- kernel 0: f32 -> bf16 weight conversion (streaming) ----------------
__global__ void moe_conv_bf16_kernel(const float* __restrict__ src,
                                     unsigned short* __restrict__ dst, size_t n8) {
  for (size_t i = blockIdx.x * (size_t)blockDim.x + threadIdx.x; i < n8;
       i += (size_t)gridDim.x * blockDim.x) {
    const float4* s = (const float4*)src + 2 * i;
    float4 a = s[0], b = s[1];
    uint4 o;
    o.x = pack_bf2(a.x, a.y);  o.y = pack_bf2(a.z, a.w);
    o.z = pack_bf2(b.x, b.y);  o.w = pack_bf2(b.z, b.w);
    ((uint4*)dst)[i] = o;
  }
}

// ---------------- kernel 1: zero output + expert counters ----------------
__global__ void moe_zero_kernel(float* __restrict__ out, int n, int* __restrict__ cnt) {
  if (blockIdx.x == 0 && threadIdx.x < NEXP) cnt[threadIdx.x] = 0;
  for (int i = blockIdx.x * blockDim.x + threadIdx.x; i < n; i += gridDim.x * blockDim.x)
    out[i] = 0.f;
}

// ---------------- kernel 2: router (one wave32 per token) ----------------
__global__ void moe_router_kernel(const float* __restrict__ x,
                                  const float* __restrict__ Wg,
                                  int* __restrict__ cnt,
                                  int* __restrict__ perm,
                                  float* __restrict__ pwt) {
  const int lane = threadIdx.x & 31;
  const int t = blockIdx.x * (blockDim.x >> 5) + (threadIdx.x >> 5);
  if (t >= T_TOK) return;

  float xv[32];
  const float* xr = x + (size_t)t * HID;
#pragma unroll
  for (int j = 0; j < 32; ++j) xv[j] = xr[lane + 32 * j];

  float logit[NEXP];
#pragma unroll
  for (int e = 0; e < NEXP; ++e) {
    const float* wr = Wg + (size_t)e * HID;
    float p = 0.f;
#pragma unroll
    for (int j = 0; j < 32; ++j) p += xv[j] * wr[lane + 32 * j];
#pragma unroll
    for (int off = 16; off > 0; off >>= 1) p += __shfl_xor(p, off, 32);
    logit[e] = p;
  }
  int i1 = 0; float l1 = logit[0];
#pragma unroll
  for (int e = 1; e < NEXP; ++e) if (logit[e] > l1) { l1 = logit[e]; i1 = e; }
  int i2 = (i1 == 0) ? 1 : 0; float l2 = logit[i2];
#pragma unroll
  for (int e = 0; e < NEXP; ++e)
    if (e != i1 && logit[e] > l2) { l2 = logit[e]; i2 = e; }
  const float wA = 1.f / (1.f + __expf(l2 - l1));
  const float wB = 1.f - wA;

  if (lane == 0) {
    int p1 = atomicAdd(&cnt[i1], 1);
    perm[i1 * T_TOK + p1] = t;  pwt[i1 * T_TOK + p1] = wA;
    int p2 = atomicAdd(&cnt[i2], 1);
    perm[i2 * T_TOK + p2] = t;  pwt[i2 * T_TOK + p2] = wB;
  }
}

// ---------------- kernel 3: fused SwiGLU FFN per expert ----------------
// LDS layout (dynamic, ~83 KB; CDNA5 WGP has 320 KB):
#define XS_STRIDE 1032                    // ushorts; 2064 B pitch (16B aligned, bank-staggered)
#define HS_STRIDE (ICHUNK + 8)            // 264 ushorts; 528 B pitch (16B aligned)
#define XS_BYTES  (MT * XS_STRIDE * 2)    // 66048
#define HS_BYTES  (MT * HS_STRIDE * 2)    // 16896
#define SMEM_BYTES (XS_BYTES + HS_BYTES + 256)

// B fragment: 16 contiguous bf16 per lane at row (n), col k0 = 16*half
template <bool BF16W>
__device__ __forceinline__ void load_B(ABFrag& b, const void* wrow, int kb) {
  if constexpr (BF16W) {
    const unsigned short* r = (const unsigned short*)wrow;
    b.q[0] = *(const uint4*)&r[kb];
    b.q[1] = *(const uint4*)&r[kb + 8];
  } else {
    const float* r = (const float*)wrow;
#pragma unroll
    for (int v = 0; v < 8; ++v) {
      float2 f = *(const float2*)&r[kb + 2 * v];
      b.u[v] = pack_bf2(f.x, f.y);
    }
  }
}

// A fragment: 16 contiguous values per lane: [16*half, +16)
__device__ __forceinline__ void load_A_lds(ABFrag& a, const unsigned short* row, int kbase, int half) {
  a.q[0] = *(const uint4*)&row[kbase + 16 * half];
  a.q[1] = *(const uint4*)&row[kbase + 16 * half + 8];
}

template <bool BF16W>
__global__ void __launch_bounds__(32 * NWAVE)
moe_ffn_kernel(const float* __restrict__ x,
               const void* __restrict__ w1p,
               const void* __restrict__ w3p,
               const void* __restrict__ w2p,
               const int* __restrict__ cnt,
               const int* __restrict__ perm,
               const float* __restrict__ pwt,
               float* __restrict__ out) {
  const int e = blockIdx.y;
  const int nrows = cnt[e];
  const int rowbase = blockIdx.x * MT;
  if (rowbase >= nrows) return;

  extern __shared__ char smem[];
  unsigned short* Xs = (unsigned short*)smem;                    // MT x 1024 bf16 X tile
  unsigned short* Hs = (unsigned short*)(smem + XS_BYTES);       // MT x ICHUNK bf16 h-chunk
  int*   toks = (int*)(smem + XS_BYTES + HS_BYTES);
  float* wts  = (float*)(smem + XS_BYTES + HS_BYTES + 128);

  const int tid  = threadIdx.x;
  const int lane = tid & 31;
  const int wid  = tid >> 5;          // 16 waves
  const int half = lane >> 4;         // 0/1
  const int nlo  = lane & 15;

  if (tid < MT) {
    int r  = rowbase + tid;
    int rc = r < nrows ? r : (nrows - 1);
    toks[tid] = perm[e * T_TOK + rc];
    wts[tid]  = (r < nrows) ? pwt[e * T_TOK + rc] : 0.f;
  }
  __syncthreads();

  // gather MT token rows (f32) -> bf16 LDS; 16 threads per row, coalesced float4
  {
    const int m   = tid >> 4;     // 0..31
    const int l16 = tid & 15;
    const float4* xr = (const float4*)(x + (size_t)toks[m] * HID);
#pragma unroll
    for (int v = 0; v < 16; ++v) {
      float4 f = xr[l16 + 16 * v];
      int col = (l16 + 16 * v) * 4;
      uint2 o = { pack_bf2(f.x, f.y), pack_bf2(f.z, f.w) };
      *(uint2*)&Xs[m * XS_STRIDE + col] = o;
    }
  }
  __syncthreads();

  // accumulators: this wave owns out rows 0..31 x H columns [wid*64, wid*64+64)
  f32x8 accA[4], accB[4];
#pragma unroll
  for (int t = 0; t < 4; ++t) { accA[t] = {}; accB[t] = {}; }

  const size_t wbase = (size_t)e * IDIM * HID;   // w1/w3: [E][I][H], w2: [E][H][I]
  const int esz = BF16W ? 2 : 4;
  const unsigned short* xsrow0 = &Xs[(size_t)nlo * XS_STRIDE];         // A rows 0..15
  const unsigned short* xsrow1 = &Xs[(size_t)(16 + nlo) * XS_STRIDE];  // A rows 16..31
  const unsigned short* hsrow0 = &Hs[(size_t)nlo * HS_STRIDE];
  const unsigned short* hsrow1 = &Hs[(size_t)(16 + nlo) * HS_STRIDE];

  for (int ic = 0; ic < IDIM / ICHUNK; ++ic) {
    const int ibase = ic * ICHUNK;
    const int iw = ibase + wid * 16;             // this wave's 16-wide i-subtile
    const char* w1row = (const char*)w1p + (wbase + (size_t)(iw + nlo) * HID) * esz;
    const char* w3row = (const char*)w3p + (wbase + (size_t)(iw + nlo) * HID) * esz;
    if (ibase + ICHUNK < IDIM) {                 // stream next chunk (global_prefetch_b8)
      __builtin_prefetch(w1row + (size_t)ICHUNK * HID * esz, 0, 0);
      __builtin_prefetch(w3row + (size_t)ICHUNK * HID * esz, 0, 0);
    }

    // h-phase: each B fragment feeds both M-halves (2x reuse)
    f32x8 h1a = {}, h1b = {}, h3a = {}, h3b = {};
#pragma unroll 4
    for (int kk = 0; kk < 32; ++kk) {            // contract over H in K=32 steps
      const int kbase = kk * 32;
      const int kb = kbase + 16 * half;
      ABFrag a0, a1, b1, b3;
      load_A_lds(a0, xsrow0, kbase, half);
      load_A_lds(a1, xsrow1, kbase, half);
      load_B<BF16W>(b1, w1row, kb);
      load_B<BF16W>(b3, w3row, kb);
      h1a = __builtin_amdgcn_wmma_f32_16x16x32_bf16(false, a0.v, false, b1.v, (short)0, h1a, false, false);
      h1b = __builtin_amdgcn_wmma_f32_16x16x32_bf16(false, a1.v, false, b1.v, (short)0, h1b, false, false);
      h3a = __builtin_amdgcn_wmma_f32_16x16x32_bf16(false, a0.v, false, b3.v, (short)0, h3a, false, false);
      h3b = __builtin_amdgcn_wmma_f32_16x16x32_bf16(false, a1.v, false, b3.v, (short)0, h3b, false, false);
    }
    // SwiGLU activation -> bf16 h-chunk in LDS (C layout: M = r + 8*half, N = lane&15)
#pragma unroll
    for (int r = 0; r < 8; ++r) {
      float ga = h1a[r], gb = h1b[r];
      float va = ga * __builtin_amdgcn_rcpf(1.f + __expf(-ga)) * h3a[r];
      float vb = gb * __builtin_amdgcn_rcpf(1.f + __expf(-gb)) * h3b[r];
      Hs[(size_t)(r + 8 * half) * HS_STRIDE + wid * 16 + nlo]        = f2bf(va);
      Hs[(size_t)(16 + r + 8 * half) * HS_STRIDE + wid * 16 + nlo]   = f2bf(vb);
    }
    __syncthreads();

    // down-projection: wave owns H columns [wid*64, wid*64+64); B reused by both M-halves
#pragma unroll
    for (int t = 0; t < 4; ++t) {
      const int hcol = wid * 64 + t * 16 + nlo;
      const char* w2row = (const char*)w2p + (wbase + (size_t)hcol * IDIM + ibase) * esz;
      f32x8 ca = accA[t], cb = accB[t];
#pragma unroll
      for (int ks = 0; ks < ICHUNK / 32; ++ks) {
        ABFrag a0, a1, b;
        load_A_lds(a0, hsrow0, ks * 32, half);
        load_A_lds(a1, hsrow1, ks * 32, half);
        load_B<BF16W>(b, w2row, ks * 32 + 16 * half);
        ca = __builtin_amdgcn_wmma_f32_16x16x32_bf16(false, a0.v, false, b.v, (short)0, ca, false, false);
        cb = __builtin_amdgcn_wmma_f32_16x16x32_bf16(false, a1.v, false, b.v, (short)0, cb, false, false);
      }
      accA[t] = ca; accB[t] = cb;
    }
    __syncthreads();
  }

  // weighted scatter; each out element receives exactly 2 commutative f32 atomic adds
#pragma unroll
  for (int t = 0; t < 4; ++t) {
    const int hcol = wid * 64 + t * 16 + nlo;
#pragma unroll
    for (int r = 0; r < 8; ++r) {
      int m0 = r + 8 * half;
      int m1 = 16 + m0;
      if (rowbase + m0 < nrows)
        atomicAdd(&out[(size_t)toks[m0] * HID + hcol], wts[m0] * accA[t][r]);
      if (rowbase + m1 < nrows)
        atomicAdd(&out[(size_t)toks[m1] * HID + hcol], wts[m1] * accB[t][r]);
    }
  }
}

// ---------------- host launcher ----------------
extern "C" void kernel_launch(void* const* d_in, const int* in_sizes, int n_in,
                              void* d_out, int out_size, void* d_ws, size_t ws_size,
                              hipStream_t stream) {
  const float* x  = (const float*)d_in[0];
  const float* Wg = (const float*)d_in[1];
  const float* w1 = (const float*)d_in[2];
  const float* w3 = (const float*)d_in[3];
  const float* w2 = (const float*)d_in[4];
  float* out = (float*)d_out;

  const size_t WELEM  = (size_t)NEXP * IDIM * HID;            // 33.5M elems per tensor
  const size_t WBYTES = 3 * WELEM * sizeof(unsigned short);   // 201 MB bf16 weights
  const size_t RBYTES = 64 + (size_t)NEXP * T_TOK * 8;        // counters + perm + pwt

  const bool preconv = ws_size >= WBYTES + RBYTES;
  dim3 grid(T_TOK / MT, NEXP);

  if (preconv) {
    unsigned short* bw1 = (unsigned short*)d_ws;
    unsigned short* bw3 = bw1 + WELEM;
    unsigned short* bw2 = bw3 + WELEM;
    int*   cnt  = (int*)((char*)d_ws + WBYTES);
    int*   perm = cnt + 16;
    float* pwt  = (float*)(perm + NEXP * T_TOK);

    (void)hipFuncSetAttribute((const void*)moe_ffn_kernel<true>,
                              hipFuncAttributeMaxDynamicSharedMemorySize, SMEM_BYTES);
    moe_conv_bf16_kernel<<<4096, 256, 0, stream>>>(w1, bw1, WELEM / 8);
    moe_conv_bf16_kernel<<<4096, 256, 0, stream>>>(w3, bw3, WELEM / 8);
    moe_conv_bf16_kernel<<<4096, 256, 0, stream>>>(w2, bw2, WELEM / 8);
    moe_zero_kernel<<<4096, 256, 0, stream>>>(out, T_TOK * HID, cnt);
    moe_router_kernel<<<T_TOK / 8, 256, 0, stream>>>(x, Wg, cnt, perm, pwt);
    moe_ffn_kernel<true><<<grid, 32 * NWAVE, SMEM_BYTES, stream>>>(
        x, bw1, bw3, bw2, cnt, perm, pwt, out);
  } else {
    int*   cnt  = (int*)d_ws;
    int*   perm = cnt + 16;
    float* pwt  = (float*)(perm + NEXP * T_TOK);

    (void)hipFuncSetAttribute((const void*)moe_ffn_kernel<false>,
                              hipFuncAttributeMaxDynamicSharedMemorySize, SMEM_BYTES);
    moe_zero_kernel<<<4096, 256, 0, stream>>>(out, T_TOK * HID, cnt);
    moe_router_kernel<<<T_TOK / 8, 256, 0, stream>>>(x, Wg, cnt, perm, pwt);
    moe_ffn_kernel<false><<<grid, 32 * NWAVE, SMEM_BYTES, stream>>>(
        x, w1, w3, w2, cnt, perm, pwt, out);
  }
}
